// QuadrangleAttention_65558380806567
// MI455X (gfx1250) — matrix operand order repaired and checked by reading, not compiled, not run on hardware
//
#include <hip/hip_runtime.h>
#include <hip/hip_bf16.h>

typedef _Float16 f16;
typedef __attribute__((ext_vector_type(16))) _Float16 v16h;
typedef __attribute__((ext_vector_type(8)))  float    v8f;

#define B_    8
#define Hh    112
#define Ww    112
#define Cc    256
#define NHh   8
#define HDd   32
#define WNn   16
#define NPIX  (Hh * Ww)            // 12544
#define MROWS (B_ * NPIX)          // 100352
#define NWH   (B_ * NHh * WNn * WNn) // 16384
#define SCALE_Q 0.17677669529663687f  // 1/sqrt(32)
#define CLAMBDA 0.01f
#define AWAVES 4                     // waves per attention block

// ---------------------------------------------------------------------------
// f32 -> f16 cast (grid-stride)
// ---------------------------------------------------------------------------
__global__ void cast_kernel(const float* __restrict__ src, f16* __restrict__ dst, long n) {
    long i = (long)blockIdx.x * blockDim.x + threadIdx.x;
    const long stride = (long)gridDim.x * blockDim.x;
    for (; i < n; i += stride) dst[i] = (f16)src[i];
}

// ---------------------------------------------------------------------------
// WMMA GEMM: C[m,n] = sum_k A[m,k]*B[n,k] (+bias[n])
// A: [M,K] f16 row-major, B: [N,K] f16 row-major.
// One wave computes a 32(M) x 64(N) tile (2 m-tiles share the B fragments);
// 4 waves / block over M.
// A-fragment: row (lane&15), halves {k0+kh*16 .. +15} (contiguous, same as B).
// B-fragment: row n (lane&15 of tile), halves {k0+kh*16 .. +15} (contiguous).
// ---------------------------------------------------------------------------
__global__ __launch_bounds__(128) void gemm_wmma_kernel(
    const f16* __restrict__ A, const f16* __restrict__ Bm,
    const float* __restrict__ bias, float* __restrict__ outF, f16* __restrict__ outH,
    int M, int N, int K)
{
    const int lane = threadIdx.x & 31;
    const int wave = threadIdx.x >> 5;
    const int m0 = (blockIdx.x * 4 + wave) * 32;
    const int n0 = blockIdx.y * 64;
    if (m0 >= M) return;
    const int kh = lane >> 4;   // 0 or 1
    const int ml = lane & 15;

    v8f acc[2][4] = {};
    const f16* arow0 = A + (long)(m0 + ml) * K;
    const f16* arow1 = A + (long)(m0 + 16 + ml) * K;
    for (int k0 = 0; k0 < K; k0 += 32) {
        v16h af0, af1;
        const f16* ap0 = arow0 + k0 + kh * 16;
        const f16* ap1 = arow1 + k0 + kh * 16;
        #pragma unroll
        for (int i = 0; i < 8; ++i) {
            af0[i] = ap0[i]; af0[8 + i] = ap0[8 + i];
            af1[i] = ap1[i]; af1[8 + i] = ap1[8 + i];
        }
        // prefetch next k-step of the streaming A operand (global_prefetch_b8)
        __builtin_prefetch(ap0 + 32, 0, 3);
        __builtin_prefetch(ap1 + 32, 0, 3);
        const f16* bbase = Bm + (long)(n0 + ml) * K + k0 + kh * 16;
        v16h b0 = *(const v16h*)(bbase);
        v16h b1 = *(const v16h*)(bbase + 16L * K);
        v16h b2 = *(const v16h*)(bbase + 32L * K);
        v16h b3 = *(const v16h*)(bbase + 48L * K);
        acc[0][0] = __builtin_amdgcn_wmma_f32_16x16x32_f16(false, af0, false, b0, (short)0, acc[0][0], false, false);
        acc[1][0] = __builtin_amdgcn_wmma_f32_16x16x32_f16(false, af1, false, b0, (short)0, acc[1][0], false, false);
        acc[0][1] = __builtin_amdgcn_wmma_f32_16x16x32_f16(false, af0, false, b1, (short)0, acc[0][1], false, false);
        acc[1][1] = __builtin_amdgcn_wmma_f32_16x16x32_f16(false, af1, false, b1, (short)0, acc[1][1], false, false);
        acc[0][2] = __builtin_amdgcn_wmma_f32_16x16x32_f16(false, af0, false, b2, (short)0, acc[0][2], false, false);
        acc[1][2] = __builtin_amdgcn_wmma_f32_16x16x32_f16(false, af1, false, b2, (short)0, acc[1][2], false, false);
        acc[0][3] = __builtin_amdgcn_wmma_f32_16x16x32_f16(false, af0, false, b3, (short)0, acc[0][3], false, false);
        acc[1][3] = __builtin_amdgcn_wmma_f32_16x16x32_f16(false, af1, false, b3, (short)0, acc[1][3], false, false);
    }
    #pragma unroll
    for (int mh = 0; mh < 2; ++mh) {
        #pragma unroll
        for (int nt = 0; nt < 4; ++nt) {
            const int n = n0 + nt * 16 + ml;
            const float bv = bias ? bias[n] : 0.0f;
            #pragma unroll
            for (int r = 0; r < 8; ++r) {
                const long m = m0 + mh * 16 + kh * 8 + r;
                const float v = acc[mh][nt][r] + bv;
                if (outF) outF[m * (long)N + n] = v;
                else      outH[m * (long)N + n] = (f16)v;
            }
        }
    }
}

// ---------------------------------------------------------------------------
// 7x7 average pool of x (b,N,C) -> pooled[(b*16+wy)*16+wx][c]
// ---------------------------------------------------------------------------
__global__ void pool_kernel(const float* __restrict__ x, float* __restrict__ pooled) {
    const int c = threadIdx.x;          // 0..255
    const int wpos = blockIdx.x;        // 0..2047 : (b, wy, wx)
    const int wx = wpos & 15, wy = (wpos >> 4) & 15, b = wpos >> 8;
    float s = 0.0f;
    for (int iy = 0; iy < 7; ++iy)
        for (int ix = 0; ix < 7; ++ix)
            s += x[((long)b * NPIX + (long)(wy * 7 + iy) * Ww + wx * 7 + ix) * Cc + c];
    pooled[(long)wpos * Cc + c] = s * (1.0f / 49.0f);
}

// ---------------------------------------------------------------------------
// samp[wpos][o] = b_t[o] + sum_c W_t[o,c] * leaky(pooled[wpos][c])
// ---------------------------------------------------------------------------
__global__ void samp_kernel(const float* __restrict__ pooled, const float* __restrict__ Wt,
                            const float* __restrict__ bt, float* __restrict__ samp) {
    const int o = threadIdx.x;
    const int wpos = blockIdx.x;
    if (o >= 72) return;
    const float* p = pooled + (long)wpos * Cc;
    float s = bt[o];
    for (int c = 0; c < Cc; ++c) {
        float pv = p[c];
        pv = (pv >= 0.0f) ? pv : 0.01f * pv;
        s += Wt[o * Cc + c] * pv;
    }
    samp[(long)wpos * 72 + o] = s;
}

// ---------------------------------------------------------------------------
// Per window-head affine/perspective transform of the 49 base coords.
// id = ((b*8 + n)*16 + wy)*16 + wx ; outputs twc2[id][0..48] (x) [49..97] (y)
// ---------------------------------------------------------------------------
__global__ void affine_kernel(const float* __restrict__ samp, float* __restrict__ twc2) {
    const int id = blockIdx.x * blockDim.x + threadIdx.x;
    if (id >= NWH) return;
    const int wx = id & 15, wy = (id >> 4) & 15, n = (id >> 8) & 7, b = id >> 11;
    const int wpos = (b * 16 + wy) * 16 + wx;
    const float* sp = samp + (long)wpos * 72 + n * 9;
    const float offx = sp[0] * (1.0f / 16.0f);
    const float offy = sp[1] * (1.0f / 16.0f);
    const float sc0 = sp[2] + 1.0f, sc1 = sp[3] + 1.0f;
    const float sh0 = sp[4], sh1 = sp[5];
    const float pr0 = sp[6], pr1 = sp[7];
    const float rot = sp[8];
    const float cr = cosf(rot), sr = sinf(rot);
    // basic = rot @ shear @ scale
    const float b00 =  cr * sc0 + sr * sh1 * sc0;
    const float b01 =  cr * sh0 * sc1 + sr * sc1;
    const float b10 = -sr * sc0 + cr * sh1 * sc0;
    const float b11 = -sr * sh0 * sc1 + cr * sc1;
    float* outp = twc2 + (long)id * 98;
    for (int s = 0; s < 49; ++s) {
        const int iy = s / 7, ix = s % 7;
        const float wcx = (float)(ix - 3) * (2.0f / 111.0f);
        const float wcy = (float)(iy - 3) * (2.0f / 111.0f);
        const float t0 = b00 * wcx + b01 * wcy + offx;
        const float t1 = b10 * wcx + b11 * wcy + offy;
        float t2 = pr0 * wcx + pr1 * wcy + 1.0f;
        if (t2 == 0.0f) t2 = 1e-6f;
        outp[s]      = t0 / t2;
        outp[49 + s] = t1 / t2;
    }
}

// ---------------------------------------------------------------------------
// Windowed attention, one wave per window-head (AWAVES waves / block).
// LDS per wave: qw/kw (union with P), S (f32 64x68), vwT, twc2, rel buffers.
// CDNA5 WGP has 320KB LDS -> 4 waves * ~30KB static LDS per block.
// ---------------------------------------------------------------------------
struct WaveSmem {
    union {
        struct { f16 qw[64][32]; f16 kw[64][32]; } a;  // pre-softmax
        f16 P[64][64];                                 // softmax probs (f16)
    } u;
    float S[64][68];   // scores (padded stride vs bank conflicts)
    f16   vwT[32][64]; // v, channel-major: vwT[d][s]
    float tw0[52];
    float tw1[52];
    float rbuf[16];    // [0..6]=rel_h(ky), [8..14]=rel_w(kx)
};

__global__ __launch_bounds__(32 * AWAVES) void attn_kernel(
    const f16* __restrict__ qk,      // [MROWS, 512] f16 : q cols 0..255, k cols 256..511
    const float* __restrict__ vsrc,  // [MROWS, 256] f32 (v_ input)
    const float* __restrict__ tws,   // [NWH, 98]
    const float* __restrict__ rph,   // [13, 32]
    const float* __restrict__ rpw,   // [13, 32]
    f16* __restrict__ aout)          // [MROWS, 256] f16
{
    __shared__ WaveSmem sm[AWAVES];
    const int lane = threadIdx.x & 31;
    const int wv = threadIdx.x >> 5;
    WaveSmem& s = sm[wv];
    const int id = blockIdx.x * AWAVES + wv;   // (b, n, wy, wx)
    const int wx = id & 15, wy = (id >> 4) & 15, hn = (id >> 8) & 7, bb = id >> 11;
    const int kh = lane >> 4, ml = lane & 15;
    const long rowbase = (long)bb * NPIX;

    // zero padded tiles
    {
        f16* zq = &s.u.a.qw[0][0];
        f16* zk = &s.u.a.kw[0][0];
        f16* zv = &s.vwT[0][0];
        for (int i = lane; i < 64 * 32; i += 32) { zq[i] = (f16)0.f; zk[i] = (f16)0.f; zv[i] = (f16)0.f; }
    }
    // load twc2
    const float* twp = tws + (long)id * 98;
    for (int i = lane; i < 49; i += 32) { s.tw0[i] = twp[i]; s.tw1[i] = twp[49 + i]; }
    __syncthreads();

    // load q window (lane = channel d)
    for (int t = 0; t < 49; ++t) {
        const long row = rowbase + (long)(wy * 7 + t / 7) * Ww + wx * 7 + t % 7;
        s.u.a.qw[t][lane] = qk[row * 512 + hn * 32 + lane];
    }

    // bilinear grid-sample of k and v at the 49 transformed points
    const float cenx = -1.0f + (float)(wx * 7 + 3) * (2.0f / 111.0f);
    const float ceny = -1.0f + (float)(wy * 7 + 3) * (2.0f / 111.0f);
    for (int t = 0; t < 49; ++t) {
        const float gx = (cenx + s.tw0[t] + 1.0f) * 55.5f;
        const float gy = (ceny + s.tw1[t] + 1.0f) * 55.5f;
        const float x0f = floorf(gx), y0f = floorf(gy);
        const int x0 = (int)x0f, y0 = (int)y0f;
        const float fx1 = gx - x0f, fx0 = 1.0f - fx1;
        const float fy1 = gy - y0f, fy0 = 1.0f - fy1;
        float ka = 0.0f, va = 0.0f;
        #pragma unroll
        for (int nb = 0; nb < 4; ++nb) {
            const int xi = x0 + (nb & 1), yi = y0 + (nb >> 1);
            const float wgt = ((nb & 1) ? fx1 : fx0) * ((nb >> 1) ? fy1 : fy0);
            const bool ok = (xi >= 0) && (xi < Ww) && (yi >= 0) && (yi < Hh);
            const int xc = min(max(xi, 0), Ww - 1);
            const int yc = min(max(yi, 0), Hh - 1);
            const long row = rowbase + (long)yc * Ww + xc;
            const float kv = (float)qk[row * 512 + 256 + hn * 32 + lane];
            const float vv = vsrc[row * 256 + hn * 32 + lane];
            const float mlt = ok ? wgt : 0.0f;
            ka += kv * mlt; va += vv * mlt;
        }
        s.u.a.kw[t][lane] = (f16)ka;
        s.vwT[lane][t]    = (f16)va;
    }
    __syncthreads();

    // S = q @ k^T  (64x64 padded, K = hd = 32)
    #pragma unroll
    for (int mt = 0; mt < 4; ++mt) {
        v16h af;
        const f16* ap = &s.u.a.qw[mt * 16 + ml][kh * 16];
        #pragma unroll
        for (int i = 0; i < 8; ++i) { af[i] = ap[i]; af[8 + i] = ap[8 + i]; }
        #pragma unroll
        for (int nt = 0; nt < 4; ++nt) {
            v16h bf = *(const v16h*)&s.u.a.kw[nt * 16 + ml][kh * 16];
            v8f c = {};
            c = __builtin_amdgcn_wmma_f32_16x16x32_f16(false, af, false, bf, (short)0, c, false, false);
            #pragma unroll
            for (int r = 0; r < 8; ++r) s.S[mt * 16 + kh * 8 + r][nt * 16 + ml] = c[r];
        }
    }
    __syncthreads();

    // scale + rel-pos + distance penalty, row by row
    for (int q = 0; q < 49; ++q) {
        const int qy = q / 7, qx = q % 7;
        if (lane < 7) {
            const float* rr = rph + (qy - lane + 6) * 32;
            float acc = 0.0f;
            for (int c = 0; c < 32; ++c) acc += (float)s.u.a.qw[q][c] * rr[c];
            s.rbuf[lane] = acc;
        } else if (lane < 14) {
            const int kx = lane - 7;
            const float* rr = rpw + (qx - kx + 6) * 32;
            float acc = 0.0f;
            for (int c = 0; c < 32; ++c) acc += (float)s.u.a.qw[q][c] * rr[c];
            s.rbuf[8 + kx] = acc;
        }
        __syncthreads();
        const float t0q = s.tw0[q], t1q = s.tw1[q];
        #pragma unroll
        for (int half = 0; half < 2; ++half) {
            const int kcol = lane + half * 32;
            float val;
            if (kcol < 49) {
                const int ky = kcol / 7, kx = kcol % 7;
                const float dx = (t0q - (float)(kx - 3) * (2.0f / 111.0f)) * 55.5f;
                const float dy = (t1q - (float)(ky - 3) * (2.0f / 111.0f)) * 55.5f;
                val = s.S[q][kcol] * SCALE_Q + s.rbuf[ky] + s.rbuf[8 + kx]
                      - CLAMBDA * sqrtf(dx * dx + dy * dy);
            } else {
                val = -1e30f;
            }
            s.S[q][kcol] = val;
        }
        __syncthreads();
    }

    // softmax rows -> P (f16, overlays qw/kw)
    #pragma unroll
    for (int rr = 0; rr < 2; ++rr) {
        const int q = lane + rr * 32;
        if (q < 49) {
            float mx = -1e30f;
            for (int c = 0; c < 64; ++c) mx = fmaxf(mx, s.S[q][c]);
            float sum = 0.0f;
            for (int c = 0; c < 64; ++c) { const float e = __expf(s.S[q][c] - mx); s.S[q][c] = e; sum += e; }
            const float inv = 1.0f / sum;
            for (int c = 0; c < 64; ++c) s.u.P[q][c] = (f16)(s.S[q][c] * inv);
        } else {
            for (int c = 0; c < 64; ++c) s.u.P[q][c] = (f16)0.f;
        }
    }
    __syncthreads();

    // O = P @ V  (M=64, N=32, K=64)
    #pragma unroll
    for (int mt = 0; mt < 4; ++mt) {
        v8f o0 = {}, o1 = {};
        #pragma unroll
        for (int kc = 0; kc < 2; ++kc) {
            v16h af;
            const f16* ap = &s.u.P[mt * 16 + ml][kc * 32 + kh * 16];
            #pragma unroll
            for (int i = 0; i < 8; ++i) { af[i] = ap[i]; af[8 + i] = ap[8 + i]; }
            v16h b0 = *(const v16h*)&s.vwT[ml][kc * 32 + kh * 16];
            v16h b1 = *(const v16h*)&s.vwT[16 + ml][kc * 32 + kh * 16];
            o0 = __builtin_amdgcn_wmma_f32_16x16x32_f16(false, af, false, b0, (short)0, o0, false, false);
            o1 = __builtin_amdgcn_wmma_f32_16x16x32_f16(false, af, false, b1, (short)0, o1, false, false);
        }
        #pragma unroll
        for (int r = 0; r < 8; ++r) {
            const int q = mt * 16 + kh * 8 + r;
            if (q < 49) {
                const long row = rowbase + (long)(wy * 7 + q / 7) * Ww + wx * 7 + q % 7;
                aout[row * 256 + hn * 32 + ml]      = (f16)o0[r];
                aout[row * 256 + hn * 32 + 16 + ml] = (f16)o1[r];
            }
        }
    }
}

// ---------------------------------------------------------------------------
extern "C" void kernel_launch(void* const* d_in, const int* in_sizes, int n_in,
                              void* d_out, int out_size, void* d_ws, size_t ws_size,
                              hipStream_t stream)
{
    (void)in_sizes; (void)n_in; (void)out_size; (void)ws_size;
    const float* x   = (const float*)d_in[0];
    const float* v_  = (const float*)d_in[1];
    const float* Wqk = (const float*)d_in[2];
    const float* Wt  = (const float*)d_in[3];
    const float* bt  = (const float*)d_in[4];
    const float* rph = (const float*)d_in[5];
    const float* rpw = (const float*)d_in[6];
    const float* Wpj = (const float*)d_in[7];
    const float* bpj = (const float*)d_in[8];
    float* out = (float*)d_out;

    char* ws = (char*)d_ws;
    auto alloc = [&](size_t bytes) { char* p = ws; ws += (bytes + 255) & ~(size_t)255; return p; };
    f16*   x_h    = (f16*)  alloc((size_t)MROWS * 256 * 2);
    f16*   wqk_h  = (f16*)  alloc((size_t)512 * 256 * 2);
    f16*   wpj_h  = (f16*)  alloc((size_t)256 * 256 * 2);
    f16*   qk_h   = (f16*)  alloc((size_t)MROWS * 512 * 2);
    f16*   ao_h   = (f16*)  alloc((size_t)MROWS * 256 * 2);
    float* pooled = (float*)alloc((size_t)2048 * 256 * 4);
    float* samp   = (float*)alloc((size_t)2048 * 72 * 4);
    float* twc2   = (float*)alloc((size_t)NWH * 98 * 4);

    // casts to f16
    cast_kernel<<<4096, 256, 0, stream>>>(x, x_h, (long)MROWS * 256);
    cast_kernel<<<512, 256, 0, stream>>>(Wqk, wqk_h, 512L * 256);
    cast_kernel<<<256, 256, 0, stream>>>(Wpj, wpj_h, 256L * 256);

    // GEMM1: qk = x @ W_qk^T   [100352, 512] f16  (32-row waves: 3136 m-tiles)
    gemm_wmma_kernel<<<dim3(784, 8), 128, 0, stream>>>(
        x_h, wqk_h, nullptr, nullptr, qk_h, MROWS, 512, 256);

    // transformation network
    pool_kernel<<<2048, 256, 0, stream>>>(x, pooled);
    samp_kernel<<<2048, 96, 0, stream>>>(pooled, Wt, bt, samp);
    affine_kernel<<<64, 256, 0, stream>>>(samp, twc2);

    // windowed attention with grid-sampled K/V
    attn_kernel<<<NWH / AWAVES, 32 * AWAVES, 0, stream>>>(qk_h, v_, twc2, rph, rpw, ao_h);

    // GEMM2: out = attn_out @ W_proj^T + b_proj   [100352, 256] f32
    gemm_wmma_kernel<<<dim3(784, 4), 128, 0, stream>>>(
        ao_h, wpj_h, bpj, out, nullptr, MROWS, 256, 256);
}
